// NBST_68161130987975
// MI455X (gfx1250) — compile-verified, matrix operand order checked by reading, not run on hardware
//
#include <hip/hip_runtime.h>
#include <math.h>

#define BB 2048
#define SS 168
#define NST 64
#define HID 64

typedef __attribute__((ext_vector_type(16))) _Float16 v16h;
typedef __attribute__((ext_vector_type(8)))  _Float16 v8h;
typedef __attribute__((ext_vector_type(8)))  float    v8f;

// ---- branch-free activations on native v_exp_f32 / v_rcp_f32 ------------
#define LOG2E 1.4426950408889634f

__device__ __forceinline__ float fast_sigm(float x) {
  // 1/(1+e^-x); exp2 saturates to inf/0 so limits are exact.
  return __builtin_amdgcn_rcpf(1.0f + __builtin_amdgcn_exp2f(-LOG2E * x));
}
__device__ __forceinline__ float fast_tanh(float x) {
  // 1 - 2/(e^{2x}+1); x->+inf => 1, x->-inf => -1.
  return fmaf(-2.0f,
              __builtin_amdgcn_rcpf(1.0f + __builtin_amdgcn_exp2f(2.0f * LOG2E * x)),
              1.0f);
}

// ---- WMMA helpers (CDNA5 wave32, 16x16x32 f16 -> f32) -------------------

__device__ __forceinline__ v8f wmma_f16(v16h a, v16h b, v8f c) {
  return __builtin_amdgcn_wmma_f32_16x16x32_f16(false, a, false, b, (short)0, c,
                                                false, false);
}

// Load a 16x32 f16 tile (A layout; B uses the same loader applied to W stored
// as [N rows][K cols]) from LDS, row stride `ld` elements.
// Per ISA 7.12.2: lanes 0-15 hold K=0..7,16..23 ; lanes 16-31 hold K=8..15,24..31.
__device__ __forceinline__ v16h ld_tile(const _Float16* p, int ld, int lane) {
  int m  = lane & 15;
  int kb = (lane >> 4) << 3;              // 0 or 8
  const _Float16* row = p + m * ld + kb;
  v8h lo = *(const v8h*)(row);            // 16B ds_load_b128
  v8h hi = *(const v8h*)(row + 16);
  v16h o;
#pragma unroll
  for (int i = 0; i < 8; i++) { o[i] = lo[i]; o[8 + i] = hi[i]; }
  return o;
}

__device__ __forceinline__ v8f bcast8(float v) {
  v8f o;
#pragma unroll
  for (int i = 0; i < 8; i++) o[i] = v;
  return o;
}

// Convert an f32 weight matrix [rows x scols] to f16 LDS [rows x ld], zero-padded.
__device__ __forceinline__ void load_w(const float* src, _Float16* dst, int rows,
                                       int scols, int ld, int tid, int nthr) {
  int tot = rows * ld;
  for (int i = tid; i < tot; i += nthr) {
    int r = i / ld, c = i - r * ld;
    float v = (c < scols) ? src[r * scols + c] : 0.0f;
    dst[i] = (_Float16)v;
  }
}

// One GRU step for a 16-row batch tile. hreg updated in place; hs stores
// deferred until all A-reads of hs this step are done.
__device__ __forceinline__ void gru_step(const _Float16* xs, int xld, int xchunks,
                                         _Float16* hs, const _Float16* wih, int wld,
                                         const _Float16* whh, const float* bRZ,
                                         const float* bXn, const float* bHn,
                                         float hreg[4][8], int lane) {
  int col = lane & 15;
  int rb  = (lane >> 4) << 3;
#pragma unroll
  for (int j = 0; j < 4; j++) {
    v8f aR = bcast8(bRZ[j]);       // r gate: x-side + h-side + both biases
    v8f aZ = bcast8(bRZ[j + 4]);   // z gate: same
    v8f aX = bcast8(bXn[j]);       // n gate x-side (kept separate)
    v8f aH = bcast8(bHn[j]);       // n gate h-side (kept separate)
#pragma unroll
    for (int kc = 0; kc < 4; kc++) {
      if (kc >= xchunks) break;
      v16h a = ld_tile(xs + kc * 32, xld, lane);
      aR = wmma_f16(a, ld_tile(wih + (j      * 16) * wld + kc * 32, wld, lane), aR);
      aZ = wmma_f16(a, ld_tile(wih + ((j + 4) * 16) * wld + kc * 32, wld, lane), aZ);
      aX = wmma_f16(a, ld_tile(wih + ((j + 8) * 16) * wld + kc * 32, wld, lane), aX);
    }
#pragma unroll
    for (int kc = 0; kc < 2; kc++) {
      v16h ah = ld_tile(hs + kc * 32, 64, lane);
      aR = wmma_f16(ah, ld_tile(whh + (j      * 16) * 64 + kc * 32, 64, lane), aR);
      aZ = wmma_f16(ah, ld_tile(whh + ((j + 4) * 16) * 64 + kc * 32, 64, lane), aZ);
      aH = wmma_f16(ah, ld_tile(whh + ((j + 8) * 16) * 64 + kc * 32, 64, lane), aH);
    }
#pragma unroll
    for (int r = 0; r < 8; r++) {
      float rg = fast_sigm(aR[r]);
      float zg = fast_sigm(aZ[r]);
      float ng = fast_tanh(aX[r] + rg * aH[r]);
      hreg[j][r] = (1.0f - zg) * ng + zg * hreg[j][r];
    }
  }
  // Commit h_new after ALL reads of hs this step (wave-private LDS, in-order).
#pragma unroll
  for (int j = 0; j < 4; j++)
#pragma unroll
    for (int r = 0; r < 8; r++)
      hs[(rb + r) * 64 + j * 16 + col] = (_Float16)hreg[j][r];
}

// LDS byte layout (dynamic shared):
//   0      : Wih0 f16 192x128  (49152)
//   49152  : Whh0 f16 192x64   (24576)
//   73728  : Wih1 f16 192x64   (24576)
//   98304  : Whh1 f16 192x64   (24576)
//   122880 : W_d1 f16 32x64    (4096, K padded 36->64)
//   126976 : W_d2 f16 64x32    (4096)
//   131072 + wave*15360 : per-wave staging
//       +0     xstg 16x128 f16 (4096)
//       +4096  hst0 16x64  f16 (2048)
//       +6144  hst1 16x64  f16 (2048)
//       +8192  cstg 16x64  f16 (2048)
//       +10240 gstg 16x32  f16 (1024)
//       +11264 attn 16x64  f32 (4096, persistent)
#define PW_STRIDE 15360
#define SMEM_BYTES (131072 + 4 * PW_STRIDE)

__global__ __launch_bounds__(128, 1) void nbst_fused(
    const float* __restrict__ local_node, const float* __restrict__ station_nodes,
    const float* __restrict__ station_features, const int* __restrict__ station_emb,
    const float* __restrict__ emb_weather, const float* __restrict__ emb_wind,
    const float* __restrict__ emb_month, const float* __restrict__ emb_day,
    const float* __restrict__ emb_hour, const float* __restrict__ W_nodes,
    const float* __restrict__ b_nodes, const float* __restrict__ W_attn,
    const float* __restrict__ b_attn, const float* __restrict__ W_d1,
    const float* __restrict__ b_d1, const float* __restrict__ W_d2,
    const float* __restrict__ b_d2, const float* __restrict__ Wih0,
    const float* __restrict__ Whh0, const float* __restrict__ bih0,
    const float* __restrict__ bhh0, const float* __restrict__ Wih1,
    const float* __restrict__ Whh1, const float* __restrict__ bih1,
    const float* __restrict__ bhh1, const float* __restrict__ W_pred,
    const float* __restrict__ b_pred, float* __restrict__ out) {
  extern __shared__ char smem[];
  _Float16* wih0 = (_Float16*)(smem);
  _Float16* whh0 = (_Float16*)(smem + 49152);
  _Float16* wih1 = (_Float16*)(smem + 73728);
  _Float16* whh1 = (_Float16*)(smem + 98304);
  _Float16* wd1  = (_Float16*)(smem + 122880);
  _Float16* wd2  = (_Float16*)(smem + 126976);

  int tid = threadIdx.x;
  int wave = tid >> 5, lane = tid & 31;
  char* pw = smem + 131072 + wave * PW_STRIDE;
  _Float16* xstg  = (_Float16*)(pw);
  _Float16* hst0  = (_Float16*)(pw + 4096);
  _Float16* hst1  = (_Float16*)(pw + 6144);
  _Float16* cstg  = (_Float16*)(pw + 8192);
  _Float16* gstg  = (_Float16*)(pw + 10240);
  float*    attnL = (float*)(pw + 11264);

  // --- one-time: weights -> f16 LDS (all 128 threads cooperate) ---
  load_w(Wih0, wih0, 192, 128, 128, tid, 128);
  load_w(Whh0, whh0, 192, 64, 64, tid, 128);
  load_w(Wih1, wih1, 192, 64, 64, tid, 128);
  load_w(Whh1, whh1, 192, 64, 64, tid, 128);
  load_w(W_d1, wd1, 32, 36, 64, tid, 128);
  load_w(W_d2, wd2, 64, 32, 32, tid, 128);

  int bbase = blockIdx.x * 64 + wave * 16;

  // --- one-time attention + softmax (VALU; negligible FLOPs) ---
  if (lane < 16) {
    int b = bbase + lane;
    float eloc = b_attn[0];
    for (int h = 0; h < 64; h++) {
      float a = b_nodes[h];
      for (int d = 0; d < 16; d++) a += local_node[b * 16 + d] * W_nodes[h * 16 + d];
      eloc += W_attn[h] * fast_tanh(a);
    }
    float mx = -1e30f;
    for (int n = 0; n < NST; n++) {
      float e = eloc;
      const float* sn = station_nodes + (b * NST + n) * 16;
      for (int h = 0; h < 64; h++) {
        float a = b_nodes[h];
        for (int d = 0; d < 16; d++) a += sn[d] * W_nodes[h * 16 + d];
        e += W_attn[64 + h] * fast_tanh(a);
      }
      attnL[lane * 64 + n] = e;
      mx = fmaxf(mx, e);
    }
    float s = 0.0f;
    for (int n = 0; n < NST; n++) {
      float v = __builtin_amdgcn_exp2f(LOG2E * (attnL[lane * 64 + n] - mx));
      attnL[lane * 64 + n] = v;
      s += v;
    }
    float inv = __builtin_amdgcn_rcpf(s);
    for (int n = 0; n < NST; n++) attnL[lane * 64 + n] *= inv;
  }
  __syncthreads();  // weights + attention visible

  int col = lane & 15, half = lane >> 4, rb = half * 8;

  // per-lane bias registers (WMMA C layout: depend only on column)
  float bRZ0[8], bRZ1[8], bXn0[4], bHn0[4], bXn1[4], bHn1[4], bD1[2], bD2[4];
#pragma unroll
  for (int j = 0; j < 8; j++) {
    bRZ0[j] = bih0[j * 16 + col] + bhh0[j * 16 + col];
    bRZ1[j] = bih1[j * 16 + col] + bhh1[j * 16 + col];
  }
#pragma unroll
  for (int j = 0; j < 4; j++) {
    bXn0[j] = bih0[(j + 8) * 16 + col];
    bHn0[j] = bhh0[(j + 8) * 16 + col];
    bXn1[j] = bih1[(j + 8) * 16 + col];
    bHn1[j] = bhh1[(j + 8) * 16 + col];
    bD2[j]  = b_d2[j * 16 + col];
  }
  bD1[0] = b_d1[col];
  bD1[1] = b_d1[16 + col];

  // zero hidden staging + cat staging (pad cols 36..63 stay zero forever)
  for (int i = lane; i < 1024; i += 32) {
    hst0[i] = (_Float16)0.0f;
    hst1[i] = (_Float16)0.0f;
    cstg[i] = (_Float16)0.0f;
  }
  float h0reg[4][8] = {}, h1reg[4][8] = {};

  // ---------------- main recurrence over S=168 steps ----------------
  for (int t = 0; t < SS; t++) {
    // stage cat = [station_features(16) | 5 x emb(4)] as 16x64 f16 (K padded)
    {
      int m = col;
      int b = bbase + m;
      const int* ei = station_emb + (b * SS + t) * 5;
      if (half == 0) {
        const float* sf = station_features + (b * SS + t) * 16;
#pragma unroll
        for (int c = 0; c < 16; c++) cstg[m * 64 + c] = (_Float16)sf[c];
        int e0 = ei[0], e1 = ei[1], e2 = ei[2] - 1, e3 = ei[3] - 1;
#pragma unroll
        for (int c = 0; c < 4; c++) {
          cstg[m * 64 + 16 + c] = (_Float16)emb_weather[e0 * 4 + c];
          cstg[m * 64 + 20 + c] = (_Float16)emb_wind[e1 * 4 + c];
          cstg[m * 64 + 24 + c] = (_Float16)emb_month[e2 * 4 + c];
          cstg[m * 64 + 28 + c] = (_Float16)emb_day[e3 * 4 + c];
        }
      } else {
        int e4 = ei[4] - 1;
#pragma unroll
        for (int c = 0; c < 4; c++)
          cstg[m * 64 + 32 + c] = (_Float16)emb_hour[e4 * 4 + c];
      }
    }

    // h1 = tanh(cat @ W_d1.T + b_d1) : 16x32, K=64 (padded) -> 4 WMMA
    v8f g0 = bcast8(bD1[0]), g1 = bcast8(bD1[1]);
#pragma unroll
    for (int kc = 0; kc < 2; kc++) {
      v16h a = ld_tile(cstg + kc * 32, 64, lane);
      g0 = wmma_f16(a, ld_tile(wd1 + 0 * 64 + kc * 32, 64, lane), g0);
      g1 = wmma_f16(a, ld_tile(wd1 + 16 * 64 + kc * 32, 64, lane), g1);
    }
#pragma unroll
    for (int r = 0; r < 8; r++) {
      gstg[(rb + r) * 32 + col]      = (_Float16)fast_tanh(g0[r]);
      gstg[(rb + r) * 32 + 16 + col] = (_Float16)fast_tanh(g1[r]);
    }

    // sfe = tanh(h1 @ W_d2.T + b_d2) : 16x64, K=32 -> 4 WMMA
    // features = [attn*sfe | sfe] -> xstg 16x128 f16
#pragma unroll
    for (int j = 0; j < 4; j++) {
      v8f s = bcast8(bD2[j]);
      s = wmma_f16(ld_tile(gstg, 32, lane), ld_tile(wd2 + (j * 16) * 32, 32, lane), s);
#pragma unroll
      for (int r = 0; r < 8; r++) {
        float v  = fast_tanh(s[r]);
        float av = attnL[(rb + r) * 64 + j * 16 + col];
        xstg[(rb + r) * 128 + j * 16 + col]      = (_Float16)(av * v);
        xstg[(rb + r) * 128 + 64 + j * 16 + col] = (_Float16)v;
      }
    }

    // GRU layer 0: x = features (K=128), h = hst0   (72 WMMA)
    gru_step(xstg, 128, 4, hst0, wih0, 128, whh0, bRZ0, bXn0, bHn0, h0reg, lane);
    // GRU layer 1: x = out0 (= hst0, K=64), h = hst1 (48 WMMA)
    gru_step(hst0, 64, 2, hst1, wih1, 64, whh1, bRZ1, bXn1, bHn1, h1reg, lane);
  }

  // pred = tanh([hf0|hf1] @ W_pred.T + b_pred) -> (B,1)
  if (lane < 16) {
    int b = bbase + lane;
    float acc = b_pred[0];
    for (int k = 0; k < 64; k++) acc += (float)hst0[lane * 64 + k] * W_pred[k];
    for (int k = 0; k < 64; k++) acc += (float)hst1[lane * 64 + k] * W_pred[64 + k];
    out[b] = fast_tanh(acc);
  }
}

extern "C" void kernel_launch(void* const* d_in, const int* in_sizes, int n_in,
                              void* d_out, int out_size, void* d_ws, size_t ws_size,
                              hipStream_t stream) {
  (void)in_sizes; (void)n_in; (void)d_ws; (void)ws_size; (void)out_size;
  const float* local_node       = (const float*)d_in[0];
  const float* station_nodes    = (const float*)d_in[1];
  const float* station_features = (const float*)d_in[2];
  const int*   station_emb      = (const int*)d_in[3];
  const float* emb_weather      = (const float*)d_in[4];
  const float* emb_wind         = (const float*)d_in[5];
  const float* emb_month        = (const float*)d_in[6];
  const float* emb_day          = (const float*)d_in[7];
  const float* emb_hour         = (const float*)d_in[8];
  const float* W_nodes          = (const float*)d_in[9];
  const float* b_nodes          = (const float*)d_in[10];
  const float* W_attn           = (const float*)d_in[11];
  const float* b_attn           = (const float*)d_in[12];
  const float* W_d1             = (const float*)d_in[13];
  const float* b_d1             = (const float*)d_in[14];
  const float* W_d2             = (const float*)d_in[15];
  const float* b_d2             = (const float*)d_in[16];
  const float* Wih0             = (const float*)d_in[17];
  const float* Whh0             = (const float*)d_in[18];
  const float* bih0             = (const float*)d_in[19];
  const float* bhh0             = (const float*)d_in[20];
  const float* Wih1             = (const float*)d_in[21];
  const float* Whh1             = (const float*)d_in[22];
  const float* bih1             = (const float*)d_in[23];
  const float* bhh1             = (const float*)d_in[24];
  const float* W_pred           = (const float*)d_in[25];
  const float* b_pred           = (const float*)d_in[26];

  (void)hipFuncSetAttribute((const void*)nbst_fused,
                            hipFuncAttributeMaxDynamicSharedMemorySize, SMEM_BYTES);

  dim3 grid(BB / 64), block(128);
  nbst_fused<<<grid, block, SMEM_BYTES, stream>>>(
      local_node, station_nodes, station_features, station_emb, emb_weather,
      emb_wind, emb_month, emb_day, emb_hour, W_nodes, b_nodes, W_attn, b_attn,
      W_d1, b_d1, W_d2, b_d2, Wih0, Whh0, bih0, bhh0, Wih1, Whh1, bih1, bhh1,
      W_pred, b_pred, (float*)d_out);
}
